// PFNet7_13477607375224
// MI455X (gfx1250) — compile-verified, matrix operand order checked by reading, not run on hardware
//
#include <hip/hip_runtime.h>
#include <hip/hip_bf16.h>

typedef __attribute__((ext_vector_type(16))) _Float16 v16h;
typedef __attribute__((ext_vector_type(8)))  float    v8f;

#define NODES 5000
#define FEAT  12
#define NBINS 50
#define BINSZ 100
#define KNN   16
#define EPB   (BINSZ*KNN)        /* 1600 edges per chunk */
#define NEDGE (NBINS*EPB)        /* 80000 */
#define SLOPE 0.01f

__device__ __forceinline__ float lrelu(float v) { return v > 0.f ? v : SLOPE * v; }

// Permuted-K position inside the LDS activation tile. WMMA A-fragment K
// packing per lane is K = (i/8)*16 + kh*8 + i%8; storing each 32-wide K block
// with the two middle octets swapped makes every lane's 16 f16 fragment
// elements contiguous (32B) -> single v16h (ds_load_b128 x2).
__device__ __forceinline__ int kperm(int c) {
  return (c & ~31) | (c & 7) | ((c & 8) << 1) | ((c & 16) >> 1);
}

// ---------------------------------------------------------------------------
// Generic WMMA dense layer: D[16 x NPAD] = act(A[16 x KPAD] @ W + b)
// A: LDS tile, 16 rows x 128 halves, K-permuted layout (see kperm).
// W: global, fragment-ordered f16: [nt][kt][lane][16] (32B/lane, 1KB/wave,
//    fully coalesced global_load_b128).
// One wave (32 lanes) per 16-row tile. Fragment layouts per ISA 7.12.2.
// ---------------------------------------------------------------------------
template<int KPAD, int NPAD>
__device__ __forceinline__ void wmma_layer(
    const _Float16* A,
    const _Float16* __restrict__ W,
    const float* __restrict__ bias, int nvalid, bool act,
    _Float16* Aout, int aoutOff,            // optional LDS 16x128 destination
    float* __restrict__ gout, int gld, int node0)  // optional global f32 dest
{
  const int lane = threadIdx.x & 31;
  const int m16  = lane & 15;
  const int kh   = lane >> 4;
  constexpr int KT = KPAD / 32;
  for (int nt = 0; nt < NPAD / 16; ++nt) {
    v8f acc = {};
    for (int kt = 0; kt < KT; ++kt) {
      v16h a = *(const v16h*)(A + m16 * 128 + kt * 32 + kh * 16);
      v16h b = *(const v16h*)(W + (((nt * KT + kt) * 32) + lane) * 16);
      acc = __builtin_amdgcn_wmma_f32_16x16x32_f16(
          false, a, false, b, (short)0, acc, false, false);
    }
#pragma unroll
    for (int r = 0; r < 8; ++r) {
      int m   = r + kh * 8;
      int col = nt * 16 + m16;
      float v = acc[r];
      if (col < nvalid) { v += bias[col]; if (act) v = lrelu(v); }
      else              { v = 0.f; }
      if (Aout) Aout[m * 128 + kperm(aoutOff + col)] = (_Float16)v;
      if (gout && col < nvalid) {
        int node = node0 + m;
        if (node < NODES) gout[(size_t)node * gld + col] = v;
      }
    }
  }
  __syncthreads();
}

// ---------------------------------------------------------------------------
// Weight conversion + swizzle: f32 [K x N] row-major -> fragment-ordered f16
// [NP/16][KP/32][32 lanes][16 elems], zero padded.
// ---------------------------------------------------------------------------
__global__ void k_cvt(const float* __restrict__ W, _Float16* __restrict__ Wh,
                      int K, int Nn, int KP, int NP) {
  int t = blockIdx.x * blockDim.x + threadIdx.x;
  if (t >= KP * NP) return;
  int i    = t & 15;
  int lane = (t >> 4) & 31;
  int kt   = (t >> 9) % (KP / 32);
  int nt   = t / (512 * (KP / 32));
  int n  = lane & 15, kh = lane >> 4;
  int Kk = kt * 32 + (i >> 3) * 16 + kh * 8 + (i & 7);
  int col = nt * 16 + n;
  float v = (Kk < K && col < Nn) ? W[Kk * Nn + col] : 0.f;
  Wh[t] = (_Float16)v;
}

// ---------------------------------------------------------------------------
// nn1: 12 -> 125 -> 125 -> 12 (leaky on hidden). One wave per 16 nodes.
// ---------------------------------------------------------------------------
__global__ void __launch_bounds__(32)
k_nn1(const float* __restrict__ x,
      const _Float16* __restrict__ W1, const float* __restrict__ b1,
      const _Float16* __restrict__ W2, const float* __restrict__ b2,
      const _Float16* __restrict__ W3, const float* __restrict__ b3,
      float* __restrict__ h1) {
  __shared__ alignas(32) _Float16 X[16 * 128];
  __shared__ alignas(32) _Float16 H[16 * 128];
  __shared__ alignas(32) _Float16 G[16 * 128];
  const int lane  = threadIdx.x;
  const int node0 = blockIdx.x * 16;
  if (lane == 0) { __builtin_prefetch(W2, 0, 1); __builtin_prefetch(W3, 0, 1); }
  for (int t = lane; t < 16 * 128; t += 32) {
    X[t] = (_Float16)0.f; H[t] = (_Float16)0.f; G[t] = (_Float16)0.f;
  }
  __syncthreads();
  for (int t = lane; t < 16 * FEAT; t += 32) {
    int m = t / FEAT, c = t % FEAT, node = node0 + m;
    if (node < NODES) X[m * 128 + kperm(c)] = (_Float16)x[node * FEAT + c];
  }
  __syncthreads();
  wmma_layer<32, 128>(X, W1, b1, 125, true,  H, 0, nullptr, 0, node0);
  wmma_layer<128, 128>(H, W2, b2, 125, true,  G, 0, nullptr, 0, node0);
  wmma_layer<128, 16>(G, W3, b3, FEAT, false, nullptr, 0, h1, FEAT, node0);
}

// ---------------------------------------------------------------------------
// LSH bin assignment: argmax over [h1@cb[:,:25], -(h1@cb[:,:25])]
// ---------------------------------------------------------------------------
__global__ void k_bins(const float* __restrict__ h1,
                       const float* __restrict__ cb, int* __restrict__ bin) {
  int n = blockIdx.x * blockDim.x + threadIdx.x;
  if (n >= NODES) return;
  float f[FEAT];
#pragma unroll
  for (int i = 0; i < FEAT; ++i) f[i] = h1[n * FEAT + i];
  float mul[25];
  for (int j = 0; j < 25; ++j) {
    float s = 0.f;
#pragma unroll
    for (int i = 0; i < FEAT; ++i) s += f[i] * cb[i * 100 + j];
    mul[j] = s;
  }
  float best = mul[0]; int bi = 0;
  for (int j = 1; j < 25; ++j) if (mul[j] > best) { best = mul[j]; bi = j; }
  for (int j = 0; j < 25; ++j) if (-mul[j] > best) { best = -mul[j]; bi = 25 + j; }
  bin[n] = bi;
}

// ---------------------------------------------------------------------------
// Stable counting-sort rank (== stable argsort) -> perm[pos] = n
// ---------------------------------------------------------------------------
__global__ void k_sort(const int* __restrict__ bin, int* __restrict__ perm) {
  __shared__ int sb[256];
  int n = blockIdx.x * blockDim.x + threadIdx.x;
  int b = (n < NODES) ? bin[n] : 0;
  int pos = 0;
  for (int base = 0; base < NODES; base += 256) {
    __syncthreads();
    int t = threadIdx.x;
    sb[t] = (base + t < NODES) ? bin[base + t] : 0x7fffffff;
    __syncthreads();
    int lim = NODES - base; if (lim > 256) lim = 256;
    for (int m = 0; m < lim; ++m) {
      int bm = sb[m], gm = base + m;
      pos += (bm < b) || (bm == b && gm < n);
    }
  }
  if (n < NODES) perm[pos] = n;
}

// ---------------------------------------------------------------------------
// Per-chunk kNN graph: sigmoid inner products + iterative top-16.
// One block per chunk; dm kept in LDS (100x100 f32 = 40KB << 320KB WGP).
// ---------------------------------------------------------------------------
__global__ void k_edges(const float* __restrict__ h1, const int* __restrict__ perm,
                        int* __restrict__ esl, int* __restrict__ edl,
                        float* __restrict__ ew) {
  __shared__ float pts[BINSZ][FEAT];
  __shared__ float dm[BINSZ][BINSZ];
  const int p = blockIdx.x, t = threadIdx.x;
  for (int q = t; q < BINSZ * FEAT; q += blockDim.x) {
    int i = q / FEAT, c = q % FEAT;
    pts[i][c] = h1[perm[p * BINSZ + i] * FEAT + c];
  }
  __syncthreads();
  if (t < BINSZ) {
    for (int j = 0; j < BINSZ; ++j) {
      float s = 0.f;
#pragma unroll
      for (int c = 0; c < FEAT; ++c) s += pts[t][c] * pts[j][c];
      dm[t][j] = 1.f / (1.f + __expf(-s));
    }
    int ebase = (p * BINSZ + t) * KNN;
    for (int k = 0; k < KNN; ++k) {
      float best = -1.f; int bj = 0;
      for (int j = 0; j < BINSZ; ++j) {
        float v = dm[t][j];
        if (v > best) { best = v; bj = j; }  // strict >: first index on ties
      }
      dm[t][bj] = -2.f;
      esl[ebase + k] = t; edl[ebase + k] = bj; ew[ebase + k] = best;
    }
  }
}

// ---------------------------------------------------------------------------
// hw = leaky(h1) @ gcn_W   [5000 x 32]
// ---------------------------------------------------------------------------
__global__ void k_hw(const float* __restrict__ h1, const float* __restrict__ W,
                     float* __restrict__ hw) {
  int n = blockIdx.x * blockDim.x + threadIdx.x;
  if (n >= NODES) return;
  float f[FEAT];
#pragma unroll
  for (int i = 0; i < FEAT; ++i) f[i] = lrelu(h1[n * FEAT + i]);
  for (int c = 0; c < 32; ++c) {
    float s = 0.f;
#pragma unroll
    for (int k = 0; k < FEAT; ++k) s += f[k] * W[k * 32 + c];
    hw[n * 32 + c] = s;
  }
}

// ---------------------------------------------------------------------------
// GCNConv aggregation. All edges intra-chunk => degrees/norm exact per chunk.
// Gather-based (deterministic, no atomics), chunk features in LDS.
// ---------------------------------------------------------------------------
__global__ void k_gcn(const float* __restrict__ hw, const int* __restrict__ perm,
                      const int* __restrict__ esl, const int* __restrict__ edl,
                      const float* __restrict__ ewg, const float* __restrict__ bias,
                      float* __restrict__ hg) {
  __shared__ int   idx[BINSZ];
  __shared__ float HL[BINSZ][32];
  __shared__ float dis[BINSZ];
  __shared__ int   sl[EPB]; __shared__ int dl[EPB]; __shared__ float w[EPB];
  const int p = blockIdx.x, t = threadIdx.x;
  if (t < BINSZ) idx[t] = perm[p * BINSZ + t];
  __syncthreads();
  for (int q = t; q < BINSZ * 32; q += blockDim.x)
    HL[q / 32][q % 32] = hw[idx[q / 32] * 32 + (q % 32)];
  for (int q = t; q < EPB; q += blockDim.x) {
    int e = p * EPB + q;
    sl[q] = esl[e]; dl[q] = edl[e]; w[q] = ewg[e];
  }
  __syncthreads();
  if (t < BINSZ) {
    float deg = 1.f;                       // self loop weight 1
    for (int e = 0; e < EPB; ++e) if (dl[e] == t) deg += w[e];
    dis[t] = rsqrtf(deg);
  }
  __syncthreads();
  if (t < BINSZ) {
    float acc[32];
    float ds = dis[t] * dis[t];
#pragma unroll
    for (int c = 0; c < 32; ++c) acc[c] = ds * HL[t][c];
    for (int e = 0; e < EPB; ++e)
      if (dl[e] == t) {
        float s = dis[sl[e]] * w[e] * dis[t];
#pragma unroll
        for (int c = 0; c < 32; ++c) acc[c] += s * HL[sl[e]][c];
      }
    for (int c = 0; c < 32; ++c) hg[idx[t] * 32 + c] = acc[c] + bias[c];
  }
}

// ---------------------------------------------------------------------------
// GraphConv(add): leaky( (sum_j w_ij x_j) @ Wrel + brel + x @ Wroot )
// ---------------------------------------------------------------------------
__global__ void k_gc(const float* __restrict__ hg, const int* __restrict__ perm,
                     const int* __restrict__ esl, const int* __restrict__ edl,
                     const float* __restrict__ ewg,
                     const float* __restrict__ Wrel, const float* __restrict__ brel,
                     const float* __restrict__ Wroot, float* __restrict__ h2) {
  __shared__ int   idx[BINSZ];
  __shared__ float HL[BINSZ][32];
  __shared__ float WR[32 * 32];
  __shared__ float WT[32 * 32];
  __shared__ int   sl[EPB]; __shared__ int dl[EPB]; __shared__ float w[EPB];
  const int p = blockIdx.x, t = threadIdx.x;
  if (t < BINSZ) idx[t] = perm[p * BINSZ + t];
  for (int q = t; q < 32 * 32; q += blockDim.x) { WR[q] = Wrel[q]; WT[q] = Wroot[q]; }
  __syncthreads();
  for (int q = t; q < BINSZ * 32; q += blockDim.x)
    HL[q / 32][q % 32] = hg[idx[q / 32] * 32 + (q % 32)];
  for (int q = t; q < EPB; q += blockDim.x) {
    int e = p * EPB + q;
    sl[q] = esl[e]; dl[q] = edl[e]; w[q] = ewg[e];
  }
  __syncthreads();
  if (t < BINSZ) {
    float agg[32];
#pragma unroll
    for (int c = 0; c < 32; ++c) agg[c] = 0.f;
    for (int e = 0; e < EPB; ++e)
      if (dl[e] == t) {
        float s = w[e];
#pragma unroll
        for (int c = 0; c < 32; ++c) agg[c] += s * HL[sl[e]][c];
      }
    for (int c = 0; c < 32; ++c) {
      float v = brel[c];
      for (int k = 0; k < 32; ++k) v += agg[k] * WR[k * 32 + c] + HL[t][k] * WT[k * 32 + c];
      h2[idx[t] * 32 + c] = lrelu(v);
    }
  }
}

// ---------------------------------------------------------------------------
// nn2 (32->125->125->6) then nn3 (38->125->125->6); cand_ids fed back as nn3
// input cols 32..37. Fully WMMA; one wave per 16 nodes.
// ---------------------------------------------------------------------------
__global__ void __launch_bounds__(32)
k_nn23(const float* __restrict__ h2,
       const _Float16* __restrict__ A1, const float* __restrict__ ab1,
       const _Float16* __restrict__ A2, const float* __restrict__ ab2,
       const _Float16* __restrict__ A3, const float* __restrict__ ab3,
       const _Float16* __restrict__ B1, const float* __restrict__ bb1,
       const _Float16* __restrict__ B2, const float* __restrict__ bb2,
       const _Float16* __restrict__ B3, const float* __restrict__ bb3,
       float* __restrict__ out_ids, float* __restrict__ out_p4) {
  __shared__ alignas(32) _Float16 X[16 * 128];
  __shared__ alignas(32) _Float16 H[16 * 128];
  __shared__ alignas(32) _Float16 G[16 * 128];
  const int lane  = threadIdx.x;
  const int node0 = blockIdx.x * 16;
  if (lane == 0) { __builtin_prefetch(A2, 0, 1); __builtin_prefetch(B2, 0, 1); }
  for (int t = lane; t < 16 * 128; t += 32) {
    X[t] = (_Float16)0.f; H[t] = (_Float16)0.f; G[t] = (_Float16)0.f;
  }
  __syncthreads();
  for (int t = lane; t < 16 * 32; t += 32) {
    int m = t / 32, c = t % 32, node = node0 + m;
    if (node < NODES) X[m * 128 + kperm(c)] = (_Float16)h2[node * 32 + c];
  }
  __syncthreads();
  // nn2
  wmma_layer<32, 128>(X, A1, ab1, 125, true,  H, 0,  nullptr, 0, node0);
  wmma_layer<128, 128>(H, A2, ab2, 125, true,  G, 0,  nullptr, 0, node0);
  wmma_layer<128, 16>(G, A3, ab3, 6,  false, X, 32, out_ids, 6, node0);  // feed nn3
  // nn3 on concat(h2, cand_ids) padded to K=64 (cols 38..63 stay zero)
  wmma_layer<64, 128>(X, B1, bb1, 125, true,  H, 0,  nullptr, 0, node0);
  wmma_layer<128, 128>(H, B2, bb2, 125, true,  G, 0,  nullptr, 0, node0);
  wmma_layer<128, 16>(G, B3, bb3, 6,  false, nullptr, 0, out_p4, 6, node0);
}

// ---------------------------------------------------------------------------
// Pass-through outputs 3 and 4 (ygen_id, ygen)
// ---------------------------------------------------------------------------
__global__ void k_copy(const float* __restrict__ yid, const float* __restrict__ yg,
                       float* __restrict__ out) {
  int t = blockIdx.x * blockDim.x + threadIdx.x;
  if (t < NODES * 6) {
    out[2 * NODES * 6 + t] = yid[t];
    out[3 * NODES * 6 + t] = yg[t];
  }
}

// ---------------------------------------------------------------------------
extern "C" void kernel_launch(void* const* d_in, const int* in_sizes, int n_in,
                              void* d_out, int out_size, void* d_ws, size_t ws_size,
                              hipStream_t stream) {
  (void)in_sizes; (void)n_in; (void)out_size; (void)ws_size;
  const float* x       = (const float*)d_in[0];
  const float* ygen_id = (const float*)d_in[1];
  const float* ygen    = (const float*)d_in[2];
  const float* cb      = (const float*)d_in[3];
  const float* n1W1 = (const float*)d_in[4];  const float* n1b1 = (const float*)d_in[5];
  const float* n1W2 = (const float*)d_in[6];  const float* n1b2 = (const float*)d_in[7];
  const float* n1W3 = (const float*)d_in[8];  const float* n1b3 = (const float*)d_in[9];
  const float* gcnW = (const float*)d_in[10]; const float* gcnB = (const float*)d_in[11];
  const float* Wrel = (const float*)d_in[12]; const float* brel = (const float*)d_in[13];
  const float* Wroot= (const float*)d_in[14];
  const float* n2W1 = (const float*)d_in[15]; const float* n2b1 = (const float*)d_in[16];
  const float* n2W2 = (const float*)d_in[17]; const float* n2b2 = (const float*)d_in[18];
  const float* n2W3 = (const float*)d_in[19]; const float* n2b3 = (const float*)d_in[20];
  const float* n3W1 = (const float*)d_in[21]; const float* n3b1 = (const float*)d_in[22];
  const float* n3W2 = (const float*)d_in[23]; const float* n3b2 = (const float*)d_in[24];
  const float* n3W3 = (const float*)d_in[25]; const float* n3b3 = (const float*)d_in[26];
  float* out = (float*)d_out;

  // Workspace carve-up (256B aligned)
  char* base = (char*)d_ws;
  size_t off = 0;
  auto alloc = [&](size_t bytes) {
    size_t r = off; off = (off + bytes + 255) & ~(size_t)255; return base + r;
  };
  float* h1   = (float*)alloc(NODES * FEAT * sizeof(float));
  float* hw   = (float*)alloc(NODES * 32 * sizeof(float));
  float* hg   = (float*)alloc(NODES * 32 * sizeof(float));
  float* h2   = (float*)alloc(NODES * 32 * sizeof(float));
  int*   bin  = (int*)  alloc(NODES * sizeof(int));
  int*   perm = (int*)  alloc(NODES * sizeof(int));
  int*   esl  = (int*)  alloc(NEDGE * sizeof(int));
  int*   edl  = (int*)  alloc(NEDGE * sizeof(int));
  float* ew   = (float*)alloc(NEDGE * sizeof(float));
  _Float16* W1a = (_Float16*)alloc(32  * 128 * 2);
  _Float16* W1b = (_Float16*)alloc(128 * 128 * 2);
  _Float16* W1c = (_Float16*)alloc(128 * 16  * 2);
  _Float16* W2a = (_Float16*)alloc(32  * 128 * 2);
  _Float16* W2b = (_Float16*)alloc(128 * 128 * 2);
  _Float16* W2c = (_Float16*)alloc(128 * 16  * 2);
  _Float16* W3a = (_Float16*)alloc(64  * 128 * 2);
  _Float16* W3b = (_Float16*)alloc(128 * 128 * 2);
  _Float16* W3c = (_Float16*)alloc(128 * 16  * 2);

  // 1) weight conversion to fragment-ordered padded f16
  auto cvt = [&](const float* W, _Float16* Wh, int K, int Nn, int KP, int NP) {
    int tot = KP * NP;
    k_cvt<<<(tot + 255) / 256, 256, 0, stream>>>(W, Wh, K, Nn, KP, NP);
  };
  cvt(n1W1, W1a, 12, 125, 32, 128);  cvt(n1W2, W1b, 125, 125, 128, 128);
  cvt(n1W3, W1c, 125, 12, 128, 16);
  cvt(n2W1, W2a, 32, 125, 32, 128);  cvt(n2W2, W2b, 125, 125, 128, 128);
  cvt(n2W3, W2c, 125, 6, 128, 16);
  cvt(n3W1, W3a, 38, 125, 64, 128);  cvt(n3W2, W3b, 125, 125, 128, 128);
  cvt(n3W3, W3c, 125, 6, 128, 16);

  const int tiles = (NODES + 15) / 16;  // 313
  // 2) nn1 MLP (WMMA)
  k_nn1<<<tiles, 32, 0, stream>>>(x, W1a, n1b1, W1b, n1b2, W1c, n1b3, h1);
  // 3) LSH bin assignment
  k_bins<<<(NODES + 127) / 128, 128, 0, stream>>>(h1, cb, bin);
  // 4) stable sort permutation
  k_sort<<<(NODES + 255) / 256, 256, 0, stream>>>(bin, perm);
  // 5) per-chunk kNN edges
  k_edges<<<NBINS, 128, 0, stream>>>(h1, perm, esl, edl, ew);
  // 6) hw = leaky(h1) @ gcn_W
  k_hw<<<(NODES + 127) / 128, 128, 0, stream>>>(h1, gcnW, hw);
  // 7) GCNConv (per-chunk, deterministic gather)
  k_gcn<<<NBINS, 128, 0, stream>>>(hw, perm, esl, edl, ew, gcnB, hg);
  // 8) GraphConv + leaky
  k_gc<<<NBINS, 128, 0, stream>>>(hg, perm, esl, edl, ew, Wrel, brel, Wroot, h2);
  // 9) nn2 + nn3 (WMMA), writes cand_ids / cand_p4 directly to d_out
  k_nn23<<<tiles, 32, 0, stream>>>(h2,
      W2a, n2b1, W2b, n2b2, W2c, n2b3,
      W3a, n3b1, W3b, n3b2, W3c, n3b3,
      out, out + NODES * 6);
  // 10) ygen_id / ygen pass-through
  k_copy<<<(NODES * 6 + 255) / 256, 256, 0, stream>>>(ygen_id, ygen, out);
}
